// FractalConsciousnessProcessor_3977139716529
// MI455X (gfx1250) — matrix operand order repaired and checked by reading, not compiled, not run on hardware
//
#include <hip/hip_runtime.h>
#include <math.h>

// ---------------------------------------------------------------------------
// FractalConsciousnessProcessor for MI455X (gfx1250, wave32)
//   psi: [B=64, S=64, E=256] fp32, num_steps (=60) scan steps.
//   - field synthesis (20-mode sinusoid matvec) done ONCE for all steps via
//     v_wmma_f32_16x16x32_f16 using the exact cos(a+b) rank-40 factorization.
//   - per-step elementwise dynamics: 1 block per batch, LDS-staged slab,
//     deterministic wave reductions, FCI coupled across blocks via
//     kernel-boundary sync (entropy partials in workspace).
// ---------------------------------------------------------------------------

typedef __attribute__((ext_vector_type(16))) _Float16 v16h;
typedef __attribute__((ext_vector_type(8)))  float    v8f;

#define B_N       64
#define S_N       64
#define E_N       256
#define M_MODES   20
#define DT_F      0.05f
#define R_CHAOS_F 3.9f
#define CHAOS_I_F 0.3f
#define WAVE_A_F  0.1f
#define WAVE_F_F  0.5f
#define PHASE0_F  0.5236f
#define EPS_F     1e-10f
#define MINM_F    1e-8f
#define MAXM_F    10.0f
#define TWO_PI_F  6.2831853071795864f
#define LN_E_F    5.545177444479562f /* ln(256) */

__device__ __forceinline__ float wave_sum32(float v) {
#pragma unroll
  for (int off = 16; off > 0; off >>= 1) v += __shfl_xor(v, off, 32);
  return v;
}

// ---------------------------------------------------------------------------
// Kernel 1: WMMA field table.
//   fieldTab[t][e] = tanh( sum_m lam[m] * cos(2pi*(m+1)*e/256 + 0.5*t*DT) )
//   e = 16*i + j  =>  cos(theta_i + phi_j) = cosT*cosP - sinT*sinP
//   D[i][j] = sum_{k<40} A[i][k] * B_t[k][j],  K padded to 64 -> 2 WMMAs.
// 256 threads = 8 waves; wave w handles steps t = w, w+8, ...
// ---------------------------------------------------------------------------
__global__ void fcp_field_wmma(const float* __restrict__ lam,
                               float* __restrict__ fieldTab, int T) {
  const int lane = threadIdx.x & 31;
  const int wave = threadIdx.x >> 5;
  const int li   = lane & 15;   // A-row i / B-col j / D-col n
  const int half = lane >> 4;

  float lv[M_MODES];
#pragma unroll
  for (int m = 0; m < M_MODES; ++m) lv[m] = lam[m];

  // A matrix (t-independent). 16-bit A 16x32 layout:
  //   lane half=0: h<8 -> K=h, h>=8 -> K=16+(h-8);  half=1: K += 8.
  v16h a0, a1;
#pragma unroll
  for (int h = 0; h < 16; ++h) {
    const int kl = (h < 8) ? (half * 8 + h) : (16 + half * 8 + (h - 8));
#pragma unroll
    for (int c = 0; c < 2; ++c) {
      const int k = c * 32 + kl;
      float v = 0.0f;
      if (k < 2 * M_MODES) {
        const int m = k >> 1;
        const float th = TWO_PI_F * (float)(m + 1) * (float)li * (1.0f / 16.0f);
        v = (k & 1) ? (-lv[m] * sinf(th)) : (lv[m] * cosf(th));
      }
      if (c == 0) a0[h] = (_Float16)v; else a1[h] = (_Float16)v;
    }
  }

  for (int t = wave; t < T; t += 8) {     // wave-uniform loop: EXEC stays full
    const float ph_t = WAVE_F_F * ((float)t * DT_F);
    // B matrix. 16-bit B 32x16 layout: same K order as A (h<8 -> K=8*half+h, h>=8 -> K=16+8*half+(h-8)).
    v16h b0, b1;
#pragma unroll
    for (int h = 0; h < 16; ++h) {
      const int kl = (h < 8) ? (half * 8 + h) : (16 + half * 8 + (h - 8));
#pragma unroll
      for (int c = 0; c < 2; ++c) {
        const int k = c * 32 + kl;
        float v = 0.0f;
        if (k < 2 * M_MODES) {
          const int m = k >> 1;
          const float ph = TWO_PI_F * (float)(m + 1) * (float)li * (1.0f / 256.0f) + ph_t;
          v = (k & 1) ? sinf(ph) : cosf(ph);
        }
        if (c == 0) b0[h] = (_Float16)v; else b1[h] = (_Float16)v;
      }
    }
    v8f acc = {};
    acc = __builtin_amdgcn_wmma_f32_16x16x32_f16(false, a0, false, b0,
                                                 (short)0, acc, false, false);
    acc = __builtin_amdgcn_wmma_f32_16x16x32_f16(false, a1, false, b1,
                                                 (short)0, acc, false, false);
    // D layout: lane L: N=L&15; VGPR r: M = r + 8*(L>>4);  e = 16*M + N
#pragma unroll
    for (int r = 0; r < 8; ++r) {
      const int mrow = r + 8 * half;
      fieldTab[t * E_N + mrow * 16 + li] = tanhf(acc[r]);
    }
  }
}

// ---------------------------------------------------------------------------
// Kernel 2: psi0 init + cos(quaternion_phase) table + entropy partial [0].
// One block per batch b; 256 threads = 8 waves; wave handles rows s=w,w+8,...
// ---------------------------------------------------------------------------
__global__ void fcp_init(const float* __restrict__ inp,
                         const float* __restrict__ spec,
                         const float* __restrict__ qp,
                         float* __restrict__ psi,
                         float* __restrict__ cosqp,
                         float* __restrict__ entPart) {
  const int b    = blockIdx.x;
  const int tid  = threadIdx.x;
  const int lane = tid & 31;
  const int wave = tid >> 5;

  __shared__ float pk[E_N], cq[E_N], red[8], wpart[8];

  // spectral L1 norm over E (deterministic: wave reduce + fixed-order sum)
  const float sv = spec[b * E_N + tid];
  float s = wave_sum32(sv);
  if (lane == 0) red[wave] = s;
  __syncthreads();
  float specsum = 0.0f;
#pragma unroll
  for (int w = 0; w < 8; ++w) specsum += red[w];
  pk[tid] = sv / (specsum + 1e-12f);
  const float c0 = cosf(qp[b * E_N + tid]);
  cq[tid] = c0;
  cosqp[b * E_N + tid] = c0;
  __syncthreads();

  float ent_acc = 0.0f;
  for (int srow = wave; srow < S_N; srow += 8) {
    const float* row = inp + ((size_t)(b * S_N + srow)) * E_N;
    float v[8]; float rs = 0.0f;
#pragma unroll
    for (int j = 0; j < 8; ++j) {
      const int e = lane + 32 * j;
      float t = fabsf(row[e]) * pk[e] * (1.0f + 0.5f * cq[e]);
      t = fminf(fmaxf(t, MINM_F), MAXM_F);
      v[j] = t; rs += t;
    }
    rs = wave_sum32(rs);
    const float inv = 1.0f / (rs + EPS_F);
    float* orow = psi + ((size_t)(b * S_N + srow)) * E_N;
    float sy = 0.0f;
#pragma unroll
    for (int j = 0; j < 8; ++j) {
      const float y = v[j] * inv;
      v[j] = y; sy += y;
      orow[lane + 32 * j] = y;
    }
    sy = wave_sum32(sy);
    const float pinv = 1.0f / (sy + EPS_F);
    float ep = 0.0f;
#pragma unroll
    for (int j = 0; j < 8; ++j) {
      const float p = v[j] * pinv;
      ep += p * logf(p + EPS_F);
    }
    ep = wave_sum32(ep);
    if (lane == 0) ent_acc += ep;
  }
  if (lane == 0) wpart[wave] = ent_acc;
  __syncthreads();
  if (tid == 0) {
    float e = 0.0f;
#pragma unroll
    for (int w = 0; w < 8; ++w) e += wpart[w];
    entPart[b] = e;  // slot 0
  }
}

// per-row update shared by both phases of the step kernel
__device__ __forceinline__ float fcp_row_update(
    const float* __restrict__ Ur, const float* __restrict__ Cr,
    const float* __restrict__ Dr, const float* __restrict__ fldrow,
    float wadd, float Dc, float* __restrict__ outRow, int lane) {
  float v[8]; float rs = 0.0f;
#pragma unroll
  for (int j = 0; j < 8; ++j) {
    const int e = lane + 32 * j;
    const float c   = Cr[e];
    const float lap = 0.25f * Ur[e] + 0.25f * Dr[e] - 0.5f * c;
    float x = fminf(fmaxf(c, EPS_F), 1.0f);   // 1-1e-10 == 1.0f in fp32
#pragma unroll
    for (int it = 0; it < 5; ++it) x = R_CHAOS_F * x * (1.0f - x);
    const float fld = fldrow[e] + wadd;
    float pn = c + DT_F * (Dc * lap + fld * c + CHAOS_I_F * (x - c));
    pn = fminf(fmaxf(fabsf(pn), MINM_F), MAXM_F);
    v[j] = pn; rs += pn;
  }
  rs = wave_sum32(rs);
  const float inv = 1.0f / (rs + EPS_F);
  float sy = 0.0f;
#pragma unroll
  for (int j = 0; j < 8; ++j) {
    const float y = v[j] * inv;
    v[j] = y; sy += y;
    outRow[lane + 32 * j] = y;
  }
  sy = wave_sum32(sy);
  const float pinv = 1.0f / (sy + EPS_F);
  float ep = 0.0f;
#pragma unroll
  for (int j = 0; j < 8; ++j) {
    const float p = v[j] * pinv;
    ep += p * logf(p + EPS_F);
  }
  return wave_sum32(ep);
}

// ---------------------------------------------------------------------------
// Kernel 3: one dynamics step. Block = batch b. In-place psi update using a
// 33-row LDS halo slab in two phases (rows 0..31 then 32..63).
// ---------------------------------------------------------------------------
__global__ void fcp_step(float* __restrict__ psi,
                         const float* __restrict__ fieldTab,
                         const float* __restrict__ cosqp,
                         float* __restrict__ entPart,
                         int k, float* __restrict__ fci_out,
                         const int* __restrict__ nsteps) {
  if (k >= *nsteps) return;
  const int b    = blockIdx.x;
  const int tid  = threadIdx.x;
  const int lane = tid & 31;
  const int wave = tid >> 5;

  __shared__ __align__(16) float slab[33 * E_N];
  __shared__ float fldrow[E_N];
  __shared__ float wpart[8];
  __shared__ float fciS;

  // field row for this (b, t): fe[e]*cos_qp[b][e]; wadd added later
  fldrow[tid] = fieldTab[k * E_N + tid] * cosqp[b * E_N + tid];

  // FCI from previous state's entropy partials (fixed-order, deterministic)
  if (tid == 0) {
    const float* ep = entPart + (size_t)k * B_N;
    float hs = 0.0f;
    for (int i = 0; i < B_N; ++i) hs += ep[i];
    const float fci = -hs / ((float)(B_N * S_N) * LN_E_F);
    fciS = fci;
    if (b == 0) fci_out[k] = fci;
  }

  // phase 0 load: global rows 0..32 -> LDS rows 0..32
  {
    const float4* g4 = (const float4*)(psi + (size_t)b * S_N * E_N);
    float4* s4 = (float4*)slab;
#pragma unroll
    for (int j = 0; j < 9; ++j) {
      const int idx = tid + 256 * j;
      if (idx < 33 * (E_N / 4)) s4[idx] = g4[idx];
    }
  }
  __syncthreads();

  const float Dc   = fminf(fmaxf(1.0f + fciS, 0.01f), 10.0f);
  const float wadd = WAVE_A_F * sinf(WAVE_F_F * ((float)k * DT_F) + PHASE0_F);

  float ent_acc = 0.0f;

  // phase 0 compute: rows 0..31 (LDS row index == global row index)
  for (int srow = wave; srow < 32; srow += 8) {
    const int lu = (srow == 0) ? 0 : srow - 1;
    const float ep = fcp_row_update(slab + lu * E_N, slab + srow * E_N,
                                    slab + (srow + 1) * E_N, fldrow, wadd, Dc,
                                    psi + ((size_t)(b * S_N + srow)) * E_N, lane);
    if (lane == 0) ent_acc += ep;
  }
  __syncthreads();

  // halo shift: old rows 31,32 -> LDS rows 0,1
  slab[0 * E_N + tid] = slab[31 * E_N + tid];
  slab[1 * E_N + tid] = slab[32 * E_N + tid];
  __syncthreads();

  // phase 1 load: global rows 33..63 -> LDS rows 2..32
  {
    const float4* g4 = (const float4*)(psi + ((size_t)(b * S_N + 33)) * E_N);
    float4* s4 = (float4*)(slab + 2 * E_N);
#pragma unroll
    for (int j = 0; j < 8; ++j) {
      const int idx = tid + 256 * j;
      if (idx < 31 * (E_N / 4)) s4[idx] = g4[idx];
    }
  }
  __syncthreads();

  // phase 1 compute: rows 32..63, LDS row = global row - 31
  for (int srow = 32 + wave; srow < S_N; srow += 8) {
    const int lc = srow - 31;
    const int lu = srow - 32;
    const int ld = (srow == S_N - 1) ? lc : (srow - 30);
    const float ep = fcp_row_update(slab + lu * E_N, slab + lc * E_N,
                                    slab + ld * E_N, fldrow, wadd, Dc,
                                    psi + ((size_t)(b * S_N + srow)) * E_N, lane);
    if (lane == 0) ent_acc += ep;
  }

  if (lane == 0) wpart[wave] = ent_acc;
  __syncthreads();
  if (tid == 0) {
    float e = 0.0f;
#pragma unroll
    for (int w = 0; w < 8; ++w) e += wpart[w];
    entPart[(size_t)(k + 1) * B_N + b] = e;
  }
}

// ---------------------------------------------------------------------------
extern "C" void kernel_launch(void* const* d_in, const int* in_sizes, int n_in,
                              void* d_out, int out_size, void* d_ws, size_t ws_size,
                              hipStream_t stream) {
  const float* inp  = (const float*)d_in[0];  // [64,64,256]
  const float* spec = (const float*)d_in[1];  // [64,256]
  const float* qp   = (const float*)d_in[2];  // [64,256]
  const float* lam  = (const float*)d_in[3];  // [20]
  const int*   nst  = (const int*)d_in[4];    // scalar on device

  float* psi = (float*)d_out;                 // psi_final region (also state)
  const int PSIN = B_N * S_N * E_N;
  float* fci_out = psi + PSIN;                // fci_evolution tail
  const int T = out_size - PSIN;              // host-known step count (60)

  float* ws       = (float*)d_ws;
  float* fieldTab = ws;                       // T * 256
  float* cosqp    = fieldTab + (size_t)T * E_N;        // 64 * 256
  float* entPart  = cosqp + (size_t)B_N * E_N;         // (T+1) * 64

  fcp_field_wmma<<<1, 256, 0, stream>>>(lam, fieldTab, T);
  fcp_init<<<B_N, 256, 0, stream>>>(inp, spec, qp, psi, cosqp, entPart);
  for (int k = 0; k < T; ++k) {
    fcp_step<<<B_N, 256, 0, stream>>>(psi, fieldTab, cosqp, entPart, k,
                                      fci_out, nst);
  }
}
